// MambaBlock_27925877359222
// MI455X (gfx1250) — compile-verified, matrix operand order checked by reading, not run on hardware
//
#include <hip/hip_runtime.h>
#include <hip/hip_bf16.h>

typedef _Float16 h16;
typedef __attribute__((ext_vector_type(8)))  _Float16 v8h;
typedef __attribute__((ext_vector_type(16))) _Float16 v16h;
typedef __attribute__((ext_vector_type(8)))  float    v8f;

#define D_MODEL 512
#define D_INNER 1024
#define NH 16
#define HD 64
#define NS 16
#define HID 1024
#define DPROJ 2096   // 2*D_INNER + 2*NS + NH
#define BB 8
#define TT 64
#define WW 16
#define MTOK (BB*TT*WW)   // 8192 tokens

__device__ __forceinline__ float silu_(float x) { return x / (1.f + __expf(-x)); }
__device__ __forceinline__ float softplus_(float x) {
    return (x > 20.f) ? x : log1pf(__expf(x));
}

// ---------------------------------------------------------------------------
// Convert fp32 weight [K,N] -> f16 transposed [N,K] (WMMA B-operand friendly).
// ---------------------------------------------------------------------------
__global__ __launch_bounds__(256)
void wconv_kernel(const float* __restrict__ w, h16* __restrict__ wt, int K, int N) {
    int i = blockIdx.x * 256 + threadIdx.x;
    if (i >= K * N) return;
    int k = i / N, n = i % N;
    wt[(long)n * K + k] = (h16)w[i];
}

// ---------------------------------------------------------------------------
// LayerNorm over last dim D (one wave per row, wave32), emit f16.
// ---------------------------------------------------------------------------
__global__ __launch_bounds__(256)
void ln_f16_kernel(const float* __restrict__ x, const float* __restrict__ w,
                   const float* __restrict__ b, h16* __restrict__ out, int M, int D) {
    int row  = (blockIdx.x * blockDim.x + threadIdx.x) >> 5;
    int lane = threadIdx.x & 31;
    if (row >= M) return;
    const float* xr = x + (long)row * D;
    int per = D >> 5;
    float s = 0.f, s2 = 0.f;
    for (int j = 0; j < per; ++j) {
        float v = xr[j * 32 + lane];
        s += v; s2 += v * v;
    }
    for (int m = 16; m > 0; m >>= 1) {
        s  += __shfl_xor(s,  m, 32);
        s2 += __shfl_xor(s2, m, 32);
    }
    float mean = s / (float)D;
    float var  = s2 / (float)D - mean * mean;
    float inv  = rsqrtf(var + 1e-5f);
    h16* orow = out + (long)row * D;
    for (int j = 0; j < per; ++j) {
        int c = j * 32 + lane;
        orow[c] = (h16)((xr[c] - mean) * inv * w[c] + b[c]);
    }
}

// ---------------------------------------------------------------------------
// Register-blocked WMMA GEMM: each wave computes a 2x4 block of 16x16 tiles
// (32 rows x 64 cols). All 12 b128 loads of a K-step are issued before the
// 8 v_wmma_f32_16x16x32_f16, with each B fragment in its own register range
// so the scheduler can overlap loads with matrix ops via partial loadcnt
// waits. N-tail tiles are pointer-clamped (branch-free, store-guarded).
// C = A[M,K](f16) * Bt[N,K](f16) + bias (+ resid)
// ---------------------------------------------------------------------------
__global__ __launch_bounds__(256)
void gemm_wmma_kernel(const h16* __restrict__ A, const h16* __restrict__ Bt,
                      const float* __restrict__ bias, const float* __restrict__ resid,
                      float* __restrict__ C, int M, int N, int K) {
    int wid  = blockIdx.x * 8 + (threadIdx.x >> 5);
    int lane = threadIdx.x & 31;
    int ntN  = N >> 4;
    int nmg  = M >> 5;               // groups of 2 M-tiles (32 rows)
    int nng  = (ntN + 3) >> 2;       // groups of 4 N-tiles (64 cols)
    if (wid >= nmg * nng) return;
    int mg = wid / nng, ng = wid % nng;

    int r15  = lane & 15;
    int ksel = (lane >> 4) << 3;     // lanes 16..31 take K halves {8..15, 24..31}
    const h16* arow0 = A + (long)((mg << 5) + r15) * K + ksel;
    const h16* arow1 = arow0 + (long)16 * K;

    const h16* brow[4];
    #pragma unroll
    for (int j = 0; j < 4; ++j) {
        int tn = (ng << 2) + j;
        int tc = (tn < ntN) ? tn : (ntN - 1);   // clamp tail; stores are guarded
        brow[j] = Bt + (long)((tc << 4) + r15) * K + ksel;
    }

    v8f acc0[4] = {}, acc1[4] = {};
    for (int k0 = 0; k0 < K; k0 += 32) {
        union F { v16h v; v8h h[2]; };
        // Issue every load of this K-step first, each into its own registers.
        F a0, a1, b0, b1, b2, b3;
        a0.h[0] = *(const v8h*)(arow0 + k0);
        a0.h[1] = *(const v8h*)(arow0 + k0 + 16);
        a1.h[0] = *(const v8h*)(arow1 + k0);
        a1.h[1] = *(const v8h*)(arow1 + k0 + 16);
        b0.h[0] = *(const v8h*)(brow[0] + k0);
        b0.h[1] = *(const v8h*)(brow[0] + k0 + 16);
        b1.h[0] = *(const v8h*)(brow[1] + k0);
        b1.h[1] = *(const v8h*)(brow[1] + k0 + 16);
        b2.h[0] = *(const v8h*)(brow[2] + k0);
        b2.h[1] = *(const v8h*)(brow[2] + k0 + 16);
        b3.h[0] = *(const v8h*)(brow[3] + k0);
        b3.h[1] = *(const v8h*)(brow[3] + k0 + 16);
        if (k0 + 64 <= K) {          // global_prefetch_b8 of next K-tile
            __builtin_prefetch(arow0 + k0 + 32, 0, 1);
            __builtin_prefetch(arow1 + k0 + 32, 0, 1);
        }
        acc0[0] = __builtin_amdgcn_wmma_f32_16x16x32_f16(
            false, a0.v, false, b0.v, (short)0, acc0[0], false, false);
        acc1[0] = __builtin_amdgcn_wmma_f32_16x16x32_f16(
            false, a1.v, false, b0.v, (short)0, acc1[0], false, false);
        acc0[1] = __builtin_amdgcn_wmma_f32_16x16x32_f16(
            false, a0.v, false, b1.v, (short)0, acc0[1], false, false);
        acc1[1] = __builtin_amdgcn_wmma_f32_16x16x32_f16(
            false, a1.v, false, b1.v, (short)0, acc1[1], false, false);
        acc0[2] = __builtin_amdgcn_wmma_f32_16x16x32_f16(
            false, a0.v, false, b2.v, (short)0, acc0[2], false, false);
        acc1[2] = __builtin_amdgcn_wmma_f32_16x16x32_f16(
            false, a1.v, false, b2.v, (short)0, acc1[2], false, false);
        acc0[3] = __builtin_amdgcn_wmma_f32_16x16x32_f16(
            false, a0.v, false, b3.v, (short)0, acc0[3], false, false);
        acc1[3] = __builtin_amdgcn_wmma_f32_16x16x32_f16(
            false, a1.v, false, b3.v, (short)0, acc1[3], false, false);
    }

    int rbase = (mg << 5) + ((lane >> 4) << 3);  // lanes 16..31 hold rows M=8..15
    #pragma unroll
    for (int j = 0; j < 4; ++j) {
        int tn = (ng << 2) + j;
        if (tn >= ntN) break;                    // wave-uniform tail guard
        int col = (tn << 4) + r15;
        float bv = bias ? bias[col] : 0.f;
        #pragma unroll
        for (int r = 0; r < 8; ++r) {
            long i0 = (long)(rbase + r) * N + col;
            long i1 = (long)(rbase + 16 + r) * N + col;
            float v0 = acc0[j][r] + bv;
            float v1 = acc1[j][r] + bv;
            if (resid) { v0 += resid[i0]; v1 += resid[i1]; }
            C[i0] = v0;
            C[i1] = v1;
        }
    }
}

// ---------------------------------------------------------------------------
// dt = softplus(dt_raw + dt_bias)
// ---------------------------------------------------------------------------
__global__ __launch_bounds__(256)
void dt_kernel(const float* __restrict__ proj, const float* __restrict__ dt_bias,
               float* __restrict__ dt) {
    int i = blockIdx.x * 256 + threadIdx.x;
    if (i >= MTOK * NH) return;
    int m = i >> 4, h = i & 15;
    dt[i] = softplus_(proj[(long)m * DPROJ + 2 * D_INNER + 2 * NS + h] + dt_bias[h]);
}

// ---------------------------------------------------------------------------
// Tree scan. One workgroup per (b, h); W=16 trees share 2x64KB LDS state
// (double-buffered h[W][HD][NS]) so the parent gather across W is a plain
// LDS read + workgroup barrier per step. Writes y + Dp*x in place over xh.
// ---------------------------------------------------------------------------
__global__ __launch_bounds__(256)
void tree_scan_kernel(float* __restrict__ proj, const float* __restrict__ dt,
                      const int* __restrict__ parent, const float* __restrict__ A_log,
                      const float* __restrict__ Dp) {
    extern __shared__ float smem[];
    float* h0  = smem;                       // [W*HD*NS]
    float* h1  = h0 + WW * HD * NS;          // [W*HD*NS]
    float* Bs  = h1 + WW * HD * NS;          // [W*NS]
    float* Cs  = Bs + WW * NS;               // [W*NS]
    float* dts = Cs + WW * NS;               // [W]
    float* dAs = dts + WW;                   // [W]
    int*   ps  = (int*)(dAs + WW);           // [W]

    int b = blockIdx.x >> 4;
    int h = blockIdx.x & 15;
    int tid = threadIdx.x;
    float A  = -__expf(A_log[h]);
    float Dk = Dp[h];

    for (int i = tid; i < WW * HD * NS; i += 256) h0[i] = 0.f;
    __syncthreads();

    float* hprev = h0;
    float* hcur  = h1;
    for (int t = 0; t < TT; ++t) {
        long tokbase = ((long)b * TT + t) * WW;   // token = tokbase + w
        if (tid < WW) {
            float d = dt[(tokbase + tid) * NH + h];
            dts[tid] = d;
            dAs[tid] = __expf(d * A);
            ps[tid]  = parent[tokbase + tid];
        }
        for (int i = tid; i < WW * NS; i += 256) {
            int w = i >> 4, n = i & 15;
            Bs[i] = proj[(tokbase + w) * DPROJ + 2 * D_INNER + n];
            Cs[i] = proj[(tokbase + w) * DPROJ + 2 * D_INNER + NS + n];
        }
        __syncthreads();
        #pragma unroll
        for (int j = 0; j < 4; ++j) {
            int idx = tid + j * 256;             // 0..1023 -> (w, hd)
            int w = idx >> 6, hd = idx & 63;
            long xoff = (tokbase + w) * DPROJ + D_INNER + h * HD + hd;
            float x   = proj[xoff];
            float dtw = dts[w], dAw = dAs[w];
            const float* hp = hprev + ((ps[w] * HD) + hd) * NS;
            float*       hn = hcur  + ((w * HD) + hd) * NS;
            const float* Bw = Bs + w * NS;
            const float* Cw = Cs + w * NS;
            float dtx = dtw * x;
            float y = 0.f;
            #pragma unroll
            for (int n = 0; n < NS; ++n) {
                float v = dAw * hp[n] + dtx * Bw[n];
                hn[n] = v;
                y += v * Cw[n];
            }
            proj[xoff] = y + Dk * x;             // y (+ D skip) over xh slot
        }
        __syncthreads();
        float* tsw = hprev; hprev = hcur; hcur = tsw;
    }
}

// ---------------------------------------------------------------------------
// Gated RMSNorm: out = rms(y * silu(z)) * norm_w  -> f16. One wave per row.
// ---------------------------------------------------------------------------
__global__ __launch_bounds__(256)
void gate_rms_kernel(const float* __restrict__ proj, const float* __restrict__ norm_w,
                     h16* __restrict__ out) {
    int row  = (blockIdx.x * blockDim.x + threadIdx.x) >> 5;
    int lane = threadIdx.x & 31;
    if (row >= MTOK) return;
    const float* zr = proj + (long)row * DPROJ;
    const float* yr = zr + D_INNER;
    float v[32];
    float s2 = 0.f;
    #pragma unroll
    for (int j = 0; j < 32; ++j) {
        int c = j * 32 + lane;
        float t = yr[c] * silu_(zr[c]);
        v[j] = t; s2 += t * t;
    }
    for (int m = 16; m > 0; m >>= 1) s2 += __shfl_xor(s2, m, 32);
    float inv = rsqrtf(s2 / (float)D_INNER + 1e-5f);
    h16* orow = out + (long)row * D_INNER;
    #pragma unroll
    for (int j = 0; j < 32; ++j) {
        int c = j * 32 + lane;
        orow[c] = (h16)(v[j] * inv * norm_w[c]);
    }
}

// ---------------------------------------------------------------------------
// FF gate: fh = f1 * silu(g) -> f16
// ---------------------------------------------------------------------------
__global__ __launch_bounds__(256)
void ffgate_kernel(const float* __restrict__ f, h16* __restrict__ out) {
    long i = (long)blockIdx.x * 256 + threadIdx.x;
    if (i >= (long)MTOK * HID) return;
    long m = i >> 10; int c = (int)(i & 1023);
    float f1 = f[m * (2 * HID) + c];
    float g  = f[m * (2 * HID) + HID + c];
    out[i] = (h16)(f1 * silu_(g));
}

// ---------------------------------------------------------------------------
static inline int gemm_blocks(int M, int N) {
    int ntN = N >> 4;
    int groups = (M >> 5) * ((ntN + 3) >> 2);
    return (groups + 7) / 8;
}

extern "C" void kernel_launch(void* const* d_in, const int* in_sizes, int n_in,
                              void* d_out, int out_size, void* d_ws, size_t ws_size,
                              hipStream_t stream) {
    (void)in_sizes; (void)n_in; (void)out_size; (void)ws_size;
    const float* x       = (const float*)d_in[0];
    const int*   parent  = (const int*)  d_in[1];
    const float* ln1_w   = (const float*)d_in[2];
    const float* ln1_b   = (const float*)d_in[3];
    const float* in_w    = (const float*)d_in[4];
    const float* in_b    = (const float*)d_in[5];
    const float* A_log   = (const float*)d_in[6];
    const float* dt_bias = (const float*)d_in[7];
    const float* Dp      = (const float*)d_in[8];
    const float* norm_w  = (const float*)d_in[9];
    const float* out_w   = (const float*)d_in[10];
    const float* out_b   = (const float*)d_in[11];
    const float* ln2_w   = (const float*)d_in[12];
    const float* ln2_b   = (const float*)d_in[13];
    const float* ff1_w   = (const float*)d_in[14];
    const float* ff1_b   = (const float*)d_in[15];
    const float* ff2_w   = (const float*)d_in[16];
    const float* ff2_b   = (const float*)d_in[17];
    float* out = (float*)d_out;

    char* ws = (char*)d_ws;
    size_t cur = 0;
    auto alloc = [&](size_t bytes) -> char* {
        char* p = ws + cur;
        cur = (cur + bytes + 255) & ~(size_t)255;
        return p;
    };
    float* proj = (float*)alloc((size_t)MTOK * DPROJ * 4);   // also reused for FF hidden f
    h16*   xn16 = (h16*)  alloc((size_t)MTOK * D_MODEL * 2); // xn, later yn
    h16*   g16  = (h16*)  alloc((size_t)MTOK * D_INNER * 2); // gated-rms, later ff-gated
    float* x2   = (float*)alloc((size_t)MTOK * D_MODEL * 4); // residual stream
    float* dtb  = (float*)alloc((size_t)MTOK * NH * 4);
    h16*   w1t  = (h16*)  alloc((size_t)DPROJ * D_MODEL * 2);
    h16*   w2t  = (h16*)  alloc((size_t)D_MODEL * D_INNER * 2);
    h16*   w3t  = (h16*)  alloc((size_t)(2 * HID) * D_MODEL * 2);
    h16*   w4t  = (h16*)  alloc((size_t)D_MODEL * HID * 2);

    // Weight conversions (f32 [K,N] -> f16 [N,K])
    wconv_kernel<<<(D_MODEL * DPROJ + 255) / 256, 256, 0, stream>>>(in_w,  w1t, D_MODEL, DPROJ);
    wconv_kernel<<<(D_INNER * D_MODEL + 255) / 256, 256, 0, stream>>>(out_w, w2t, D_INNER, D_MODEL);
    wconv_kernel<<<(D_MODEL * 2 * HID + 255) / 256, 256, 0, stream>>>(ff1_w, w3t, D_MODEL, 2 * HID);
    wconv_kernel<<<(HID * D_MODEL + 255) / 256, 256, 0, stream>>>(ff2_w, w4t, HID, D_MODEL);

    // LN1 -> f16
    ln_f16_kernel<<<MTOK / 8, 256, 0, stream>>>(x, ln1_w, ln1_b, xn16, MTOK, D_MODEL);

    // in_proj GEMM: [8192 x 2096] = [8192 x 512] x [512 x 2096]
    gemm_wmma_kernel<<<gemm_blocks(MTOK, DPROJ), 256, 0, stream>>>(
        xn16, w1t, in_b, nullptr, proj, MTOK, DPROJ, D_MODEL);

    // dt = softplus(dtr + bias)
    dt_kernel<<<(MTOK * NH + 255) / 256, 256, 0, stream>>>(proj, dt_bias, dtb);

    // Tree scan (one WG per (b,h); 130KB dynamic LDS on CDNA5's 320KB WGP LDS)
    {
        size_t shm = (size_t)(2 * WW * HD * NS + 2 * WW * NS + 3 * WW) * 4;
        tree_scan_kernel<<<BB * NH, 256, shm, stream>>>(proj, dtb, parent, A_log, Dp);
    }

    // Gated RMSNorm -> f16
    gate_rms_kernel<<<MTOK / 8, 256, 0, stream>>>(proj, norm_w, g16);

    // out_proj GEMM + residual(x): x2 = g16 @ out_w + out_b + x
    gemm_wmma_kernel<<<gemm_blocks(MTOK, D_MODEL), 256, 0, stream>>>(
        g16, w2t, out_b, x, x2, MTOK, D_MODEL, D_INNER);

    // LN2 -> f16 (reuse xn16)
    ln_f16_kernel<<<MTOK / 8, 256, 0, stream>>>(x2, ln2_w, ln2_b, xn16, MTOK, D_MODEL);

    // FF1 GEMM into proj region: f = yn @ ff1_w + ff1_b  [8192 x 2048]
    gemm_wmma_kernel<<<gemm_blocks(MTOK, 2 * HID), 256, 0, stream>>>(
        xn16, w3t, ff1_b, nullptr, proj, MTOK, 2 * HID, D_MODEL);

    // FF gate -> f16 (reuse g16)
    ffgate_kernel<<<(MTOK * HID + 255) / 256, 256, 0, stream>>>(proj, g16);

    // FF2 GEMM + residual(x2) -> output
    gemm_wmma_kernel<<<gemm_blocks(MTOK, D_MODEL), 256, 0, stream>>>(
        g16, w4t, ff2_b, x2, out, MTOK, D_MODEL, HID);
}
